// LearnableSphericalEncoding_2396591751824
// MI455X (gfx1250) — matrix operand order repaired and checked by reading, not compiled, not run on hardware
//
#include <hip/hip_runtime.h>
#include <hip/hip_bf16.h>

typedef __attribute__((ext_vector_type(16))) _Float16 v16h;
typedef __attribute__((ext_vector_type(2)))  _Float16 v2h;
typedef __attribute__((ext_vector_type(8)))  float    v8f;

#define N_VEC        262144
#define OUT_RES      512
#define OUT_CH       64
#define BLOCK_THREADS 256
#define WAVES_PER_BLK 8            // 256 / wave32
#define ROWS_PER_TILE 16
#define N_TILES      (N_VEC / ROWS_PER_TILE)   // 16384
#define GRID_BLOCKS  2048

// dynamic LDS layout:
//   [0, 65536)          : weights transposed, channel-major f16  wlds[ch*512 + k]
//   [65536, 65536+6144) : sphere positions f32, SoA: sx[512], sy[512], sz[512]
#define LDS_BYTES (64 * 512 * 2 + 3 * 512 * 4)

__global__ void __launch_bounds__(BLOCK_THREADS)
spherical_enc_kernel(const float* __restrict__ vec,        // [N,3]
                     const float* __restrict__ sphere_pos, // [3,512]
                     const float* __restrict__ weights,    // [512,64]
                     const float* __restrict__ sigma_p,
                     float* __restrict__ out)              // [N,64]
{
    extern __shared__ char smem[];
    _Float16* wlds = (_Float16*)smem;                 // [64][512]
    float*    sph  = (float*)(smem + 64 * 512 * 2);   // [3][512]

    const int tid = threadIdx.x;

    // ---- fill LDS: transpose+convert weights (coalesced LDS writes), copy sphere ----
    for (int j = tid; j < OUT_CH * OUT_RES; j += BLOCK_THREADS) {
        int ch = j >> 9;          // j / 512
        int k  = j & 511;
        wlds[j] = (_Float16)weights[k * OUT_CH + ch];
    }
    for (int j = tid; j < 3 * OUT_RES; j += BLOCK_THREADS)
        sph[j] = sphere_pos[j];
    __syncthreads();

    const float sigma  = sigma_p[0];
    const float inv2s2 = 0.5f / (sigma * sigma);

    const int lane   = tid & 31;
    const int wv     = tid >> 5;
    const int half16 = lane >> 4;       // which half of the wave (K split)
    const int lm     = lane & 15;       // row-in-tile (A) / channel-in-tile (B,C)

    for (int tile = blockIdx.x * WAVES_PER_BLK + wv; tile < N_TILES;
         tile += gridDim.x * WAVES_PER_BLK) {

        const int rowBase = tile * ROWS_PER_TILE;
        const int m       = rowBase + lm;           // row this lane generates probs for

        const float vx = vec[3 * m + 0];
        const float vy = vec[3 * m + 1];
        const float vz = vec[3 * m + 2];

        v8f  acc[4] = {};                            // 4 channel tiles of 16
        float partial = 0.0f;                        // per-lane partial row sum

        for (int kk = 0; kk < OUT_RES / 32; ++kk) {  // 16 K-steps of 32
            const int kb = kk * 32;

            // ---- build A fragment (prob tile, 16x32 f16) in ISA register layout ----
            v16h a;
#pragma unroll
            for (int v = 0; v < 8; ++v) {
                const int kloc0 = ((v < 4) ? (2 * v) : (2 * v + 8)) + 8 * half16;
#pragma unroll
                for (int j = 0; j < 2; ++j) {
                    const int k = kb + kloc0 + j;
                    const float sx = sph[k];
                    const float sy = sph[OUT_RES + k];
                    const float sz = sph[2 * OUT_RES + k];
                    float c = fmaf(vx, sx, fmaf(vy, sy, vz * sz));
                    c = fminf(fmaxf(c, -0.99999f), 0.99999f);
                    const float ang = acosf(c);
                    const float p   = __expf(-ang * ang * inv2s2);
                    partial += p;
                    a[2 * v + j] = (_Float16)p;
                }
            }

            // ---- load ALL four B fragments first (distinct regs -> no WAR nops,
            //      single dscnt wait), then issue the 4 WMMAs back-to-back ----
            const int bk = kb + 8 * half16;          // this lane-half's K base
            v16h b[4];
#pragma unroll
            for (int ct = 0; ct < 4; ++ct) {
                const int ch = 16 * ct + lm;
                const v2h* wp = (const v2h*)(wlds + ch * OUT_RES + bk);
#pragma unroll
                for (int v = 0; v < 8; ++v) {
                    v2h h = wp[(v < 4) ? v : (v + 4)];
                    b[ct][2 * v]     = h.x;
                    b[ct][2 * v + 1] = h.y;
                }
            }
#pragma unroll
            for (int ct = 0; ct < 4; ++ct) {
                acc[ct] = __builtin_amdgcn_wmma_f32_16x16x32_f16(
                    /*neg_a=*/false, a, /*neg_b=*/false, b[ct],
                    /*c_mod=*/(short)0, acc[ct],
                    /*reuse_a=*/false, /*reuse_b=*/false);
            }
        }

        // ---- row sums: combine lane halves, one fast rcp per row, multiply ----
        const float tot = partial + __shfl_xor(partial, 16);

#pragma unroll
        for (int r = 0; r < 8; ++r) {
            const int rloc = r + 8 * half16;                 // C layout row
            const float rs  = __shfl(tot, rloc);              // ds_bpermute broadcast
            const float inv = __builtin_amdgcn_rcpf(rs + 1e-8f);
            float* orow = out + (size_t)(rowBase + rloc) * OUT_CH + lm;
#pragma unroll
            for (int ct = 0; ct < 4; ++ct) {
                orow[16 * ct] = acc[ct][r] * inv;
            }
        }
    }
}

extern "C" void kernel_launch(void* const* d_in, const int* in_sizes, int n_in,
                              void* d_out, int out_size, void* d_ws, size_t ws_size,
                              hipStream_t stream) {
    const float* vec        = (const float*)d_in[0];   // [262144,3]
    const float* sphere_pos = (const float*)d_in[1];   // [3,512]
    const float* weights    = (const float*)d_in[2];   // [1,512,64]
    const float* sigma      = (const float*)d_in[3];   // [1]
    float* out = (float*)d_out;                        // [262144,64]

    spherical_enc_kernel<<<GRID_BLOCKS, BLOCK_THREADS, LDS_BYTES, stream>>>(
        vec, sphere_pos, weights, sigma, out);
}
